// TransformerWithLocality_8615704396236
// MI455X (gfx1250) — matrix operand order repaired and checked by reading, not compiled, not run on hardware
//
#include <hip/hip_runtime.h>

// ---------------------------------------------------------------------------
// TransformerWithLocality for gfx1250 (MI455X): bf16 WMMA everywhere GEMM-like.
// Activations live as bf16 [N][C] (pixel-major), f32 temporaries around BN.
// GEMM kernels: 2x2 register blocking, one wave = 32 pixels x 32 outputs,
// 4 WMMAs per 8 b128 loads. Dilated conv additionally stages per-tap weight
// tiles into LDS with the Tensor Data Mover (double-buffered, TENSORcnt).
// ---------------------------------------------------------------------------

typedef __attribute__((ext_vector_type(16))) __bf16 bf16x16;
typedef __attribute__((ext_vector_type(8)))  float  floatx8;

#define HW    96
#define NPIX  9216            // per image (96*96)
#define NTOT  18432           // total pixels (B=2)
#define DIMC  128

#if defined(__has_builtin)
#  if __has_builtin(__builtin_amdgcn_tensor_load_to_lds) && \
      __has_builtin(__builtin_amdgcn_s_wait_tensorcnt)
#    define USE_TDM 1
#  endif
#endif
#ifndef USE_TDM
#  define USE_TDM 0
#endif

__device__ __forceinline__ unsigned short f2bf(float f) {
  unsigned int u = __float_as_uint(f);
  unsigned int r = (u + 0x7FFFu + ((u >> 16) & 1u)) >> 16;   // RNE
  return (unsigned short)r;
}
__device__ __forceinline__ float bf2f(unsigned short h) {
  return __uint_as_float(((unsigned int)h) << 16);
}

union Frag {
  bf16x16 v;
  unsigned short u[16];
  uint4 q[2];
};

#define WMMA_BF16(A, B, C) \
  __builtin_amdgcn_wmma_f32_16x16x32_bf16(false, (A), false, (B), (short)0, (C), false, false)

#if USE_TDM
typedef unsigned int u32x4 __attribute__((ext_vector_type(4)));
typedef int          i32x4 __attribute__((ext_vector_type(4)));
typedef int          i32x8 __attribute__((ext_vector_type(8)));

// TDM: DMA one 8KB weight tile (32 o-rows x 128 i, bf16) global -> LDS.
// D# group0: count=1 | lds_addr | global_addr[56:0] | type=2 (bits 127:126).
// D# group1: data_size=2B (bits 17:16); tensor_dim0 = tile_dim0 = 4096 elems;
//            tensor_dim0_stride = 4096; 1-D tile (tile_dim1 = 0).
__device__ __forceinline__ void tdm_load_tile(unsigned lds_off, const void* gaddr) {
  const unsigned long long ga = (unsigned long long)gaddr;
  u32x4 g0 = { 1u, lds_off,
               (unsigned)(ga & 0xFFFFFFFFu),
               (unsigned)(((ga >> 32) & 0x01FFFFFFu) | 0x80000000u) };
  i32x8 g1 = { 0x00010000,          // data_size = 1 (2 bytes)
               0x10000000,          // tensor_dim0[15:0]=4096 in bits 63:48
               0x00000000,          // tensor_dim0[31:16]=0, tensor_dim1=0
               0x10000000,          // tile_dim0 = 4096 in bits 127:112
               0x00000000,          // tile_dim1 = 0, tile_dim2 = 0
               0x00001000,          // tensor_dim0_stride = 4096
               0x00000000, 0x00000000 };
  i32x4 z4 = { 0, 0, 0, 0 };
#if __clang_major__ >= 23
  i32x8 z8 = { 0, 0, 0, 0, 0, 0, 0, 0 };
  __builtin_amdgcn_tensor_load_to_lds(g0, g1, z4, z4, z8, 0);
#else
  __builtin_amdgcn_tensor_load_to_lds(g0, g1, z4, z4, 0);
#endif
}
#endif  // USE_TDM

// A fragment (16x32 bf16): lane<16 holds row M=lane, K0..7 & K16..23;
//                          lane>=16 holds row M=lane-16, K8..15 & K24..31.
// B fragment (32x16 bf16): lane<16 holds col N=lane, K0..7 & K16..23; lane>=16 K8..15 & K24..31.
// D (16x16 f32): VGPR r -> M=r (lanes 0-15) / M=r+8 (lanes 16-31), N=lane%16.

// ---------------------------------------------------------------------------
// Generic GEMM: out[m,o] = sum_k A[m,k]*W[o,k] (+bias[o]) (+res[m,o]); bf16 out.
// One wave per 32x32 tile (2x2 blocking). Cin % 32 == 0, Cout % 32 == 0.
// ---------------------------------------------------------------------------
__global__ __launch_bounds__(32)
void wmma_gemm(const unsigned short* __restrict__ A,
               const unsigned short* __restrict__ W,
               const float* __restrict__ bias,
               const unsigned short* __restrict__ res,
               unsigned short* __restrict__ out,
               int Cin, int Cout) {
  const int lane = threadIdx.x & 31;
  const int l    = lane & 15;
  const int half = lane >> 4;
  const int m0   = blockIdx.x * 32;
  const int o0   = blockIdx.y * 32;

  floatx8 acc00 = {}, acc01 = {}, acc10 = {}, acc11 = {};
  const unsigned short* arow0 = A + (size_t)(m0 + l) * Cin;
  const unsigned short* arow1 = arow0 + (size_t)16 * Cin;
  const unsigned short* wrow0 = W + (size_t)(o0 + l) * Cin;
  const unsigned short* wrow1 = wrow0 + (size_t)16 * Cin;

  for (int k0 = 0; k0 < Cin; k0 += 32) {
    Frag a0, a1, b0, b1;
    const int ka = k0 + (half ? 8 : 0);
    const int kb = k0 + (half ? 8 : 0);
    a0.q[0] = *(const uint4*)(arow0 + ka);
    a0.q[1] = *(const uint4*)(arow0 + ka + 16);
    a1.q[0] = *(const uint4*)(arow1 + ka);
    a1.q[1] = *(const uint4*)(arow1 + ka + 16);
    __builtin_prefetch(wrow0 + kb + 32, 0, 3);
    b0.q[0] = *(const uint4*)(wrow0 + kb);
    b0.q[1] = *(const uint4*)(wrow0 + kb + 16);
    b1.q[0] = *(const uint4*)(wrow1 + kb);
    b1.q[1] = *(const uint4*)(wrow1 + kb + 16);
    acc00 = WMMA_BF16(a0.v, b0.v, acc00);
    acc01 = WMMA_BF16(a0.v, b1.v, acc01);
    acc10 = WMMA_BF16(a1.v, b0.v, acc10);
    acc11 = WMMA_BF16(a1.v, b1.v, acc11);
  }

  const int oA = o0 + l;
  const int oB = oA + 16;
  const float bvA = bias ? bias[oA] : 0.f;
  const float bvB = bias ? bias[oB] : 0.f;
#pragma unroll
  for (int r = 0; r < 8; ++r) {
    const int mA = m0 + r + half * 8;
    const int mB = mA + 16;
    const size_t i00 = (size_t)mA * Cout + oA;
    const size_t i01 = (size_t)mA * Cout + oB;
    const size_t i10 = (size_t)mB * Cout + oA;
    const size_t i11 = (size_t)mB * Cout + oB;
    float v00 = acc00[r] + bvA, v01 = acc01[r] + bvB;
    float v10 = acc10[r] + bvA, v11 = acc11[r] + bvB;
    if (res) {
      v00 += bf2f(res[i00]); v01 += bf2f(res[i01]);
      v10 += bf2f(res[i10]); v11 += bf2f(res[i11]);
    }
    out[i00] = f2bf(v00); out[i01] = f2bf(v01);
    out[i10] = f2bf(v10); out[i11] = f2bf(v11);
  }
}

// Final 128->64 GEMM scattering f32 results to NCHW d_out (2x2 blocking).
__global__ __launch_bounds__(32)
void wmma_gemm_out(const unsigned short* __restrict__ A,
                   const unsigned short* __restrict__ W,
                   const float* __restrict__ bias,
                   float* __restrict__ out,
                   int Cin, int Cout) {
  const int lane = threadIdx.x & 31;
  const int l    = lane & 15;
  const int half = lane >> 4;
  const int m0   = blockIdx.x * 32;
  const int o0   = blockIdx.y * 32;

  floatx8 acc00 = {}, acc01 = {}, acc10 = {}, acc11 = {};
  const unsigned short* arow0 = A + (size_t)(m0 + l) * Cin;
  const unsigned short* arow1 = arow0 + (size_t)16 * Cin;
  const unsigned short* wrow0 = W + (size_t)(o0 + l) * Cin;
  const unsigned short* wrow1 = wrow0 + (size_t)16 * Cin;

  for (int k0 = 0; k0 < Cin; k0 += 32) {
    Frag a0, a1, b0, b1;
    const int ka = k0 + (half ? 8 : 0);
    const int kb = k0 + (half ? 8 : 0);
    a0.q[0] = *(const uint4*)(arow0 + ka);
    a0.q[1] = *(const uint4*)(arow0 + ka + 16);
    a1.q[0] = *(const uint4*)(arow1 + ka);
    a1.q[1] = *(const uint4*)(arow1 + ka + 16);
    b0.q[0] = *(const uint4*)(wrow0 + kb);
    b0.q[1] = *(const uint4*)(wrow0 + kb + 16);
    b1.q[0] = *(const uint4*)(wrow1 + kb);
    b1.q[1] = *(const uint4*)(wrow1 + kb + 16);
    acc00 = WMMA_BF16(a0.v, b0.v, acc00);
    acc01 = WMMA_BF16(a0.v, b1.v, acc01);
    acc10 = WMMA_BF16(a1.v, b0.v, acc10);
    acc11 = WMMA_BF16(a1.v, b1.v, acc11);
  }

  const int oA = o0 + l;
  const int oB = oA + 16;
  const float bvA = bias[oA];
  const float bvB = bias[oB];
#pragma unroll
  for (int r = 0; r < 8; ++r) {
    const int mA  = m0 + r + half * 8;
    const int mB  = mA + 16;
    const int bA  = mA / NPIX, rA = mA % NPIX;
    const int bB  = mB / NPIX, rB = mB % NPIX;
    out[((size_t)bA * Cout + oA) * NPIX + rA] = acc00[r] + bvA;
    out[((size_t)bA * Cout + oB) * NPIX + rA] = acc01[r] + bvB;
    out[((size_t)bB * Cout + oA) * NPIX + rB] = acc10[r] + bvA;
    out[((size_t)bB * Cout + oB) * NPIX + rB] = acc11[r] + bvB;
  }
}

// ---------------------------------------------------------------------------
// Dilated 3x3 conv as implicit GEMM with 2x2 blocking. 8 waves (256 pixels)
// per block share one 32-output weight slice. Per tap, the 8KB weight tile is
// DMA'd into LDS by the Tensor Data Mover (wave 0), double-buffered so the
// DMA for tap+1 overlaps the 16 WMMAs of tap; all waves read B from LDS.
// Falls back to direct global B loads if the TDM builtin is unavailable.
// Output f32 (pre-BN). Conv bias omitted: cancels under train-mode BN.
// ---------------------------------------------------------------------------
__global__ __launch_bounds__(256)
void wmma_dconv(const unsigned short* __restrict__ A,
                const unsigned short* __restrict__ Wt,
                float* __restrict__ out, int dil) {
#if USE_TDM
  __shared__ __align__(16) unsigned short smem[2][32 * DIMC];
#endif
  const int tid  = threadIdx.x;
  const int wave = tid >> 5;
  const int lane = tid & 31;
  const int l    = lane & 15;
  const int half = lane >> 4;
  const int m0   = (blockIdx.x * 8 + wave) * 32;  // 32 px, one row (96%32==0)
  const int o0   = blockIdx.y * 32;
  const int bimg = m0 / NPIX;
  const int rem  = m0 % NPIX;
  const int y    = rem / HW;
  const int x0   = rem % HW;

#if USE_TDM
  if (tid < 32) {
    tdm_load_tile((unsigned)(size_t)&smem[0][0], Wt + (size_t)o0 * DIMC);
    __builtin_amdgcn_s_wait_tensorcnt(0);
  }
  __syncthreads();
#endif

  floatx8 acc00 = {}, acc01 = {}, acc10 = {}, acc11 = {};
  for (int tap = 0; tap < 9; ++tap) {
#if USE_TDM
    const int cur = tap & 1;
    if (tid < 32 && tap + 1 < 9)
      tdm_load_tile((unsigned)(size_t)&smem[cur ^ 1][0],
                    Wt + ((size_t)(tap + 1) * DIMC + o0) * DIMC);
    const unsigned short* wrow0 = &smem[cur][0] + (size_t)l * DIMC;
#else
    const unsigned short* wrow0 = Wt + ((size_t)tap * DIMC + o0 + l) * DIMC;
#endif
    const unsigned short* wrow1 = wrow0 + (size_t)16 * DIMC;

    const int dy  = tap / 3 - 1;
    const int dx  = tap % 3 - 1;
    const int ys  = y + dy * dil;
    const int xs0 = x0 + l + dx * dil;
    const int xs1 = xs0 + 16;
    const bool yok = (ys >= 0) & (ys < HW);
    const bool ok0 = yok & (xs0 >= 0) & (xs0 < HW);
    const bool ok1 = yok & (xs1 >= 0) & (xs1 < HW);
    const unsigned short* arow0 = A + (size_t)(bimg * NPIX + ys * HW + xs0) * DIMC;
    const unsigned short* arow1 = arow0 + (size_t)16 * DIMC;

#pragma unroll
    for (int k0 = 0; k0 < DIMC; k0 += 32) {
      Frag a0, a1, b0, b1;
      const int ka = k0 + (half ? 8 : 0);
      const int kb = k0 + (half ? 8 : 0);
      if (ok0) {
        a0.q[0] = *(const uint4*)(arow0 + ka);
        a0.q[1] = *(const uint4*)(arow0 + ka + 16);
      } else {
        a0.q[0] = make_uint4(0u, 0u, 0u, 0u);
        a0.q[1] = make_uint4(0u, 0u, 0u, 0u);
      }
      if (ok1) {
        a1.q[0] = *(const uint4*)(arow1 + ka);
        a1.q[1] = *(const uint4*)(arow1 + ka + 16);
      } else {
        a1.q[0] = make_uint4(0u, 0u, 0u, 0u);
        a1.q[1] = make_uint4(0u, 0u, 0u, 0u);
      }
      b0.q[0] = *(const uint4*)(wrow0 + kb);
      b0.q[1] = *(const uint4*)(wrow0 + kb + 16);
      b1.q[0] = *(const uint4*)(wrow1 + kb);
      b1.q[1] = *(const uint4*)(wrow1 + kb + 16);
      acc00 = WMMA_BF16(a0.v, b0.v, acc00);
      acc01 = WMMA_BF16(a0.v, b1.v, acc01);
      acc10 = WMMA_BF16(a1.v, b0.v, acc10);
      acc11 = WMMA_BF16(a1.v, b1.v, acc11);
    }
#if USE_TDM
    if (tid < 32 && tap + 1 < 9) __builtin_amdgcn_s_wait_tensorcnt(0);
    __syncthreads();
#endif
  }

  const int oA = o0 + l;
  const int oB = oA + 16;
#pragma unroll
  for (int r = 0; r < 8; ++r) {
    const int mA = m0 + r + half * 8;
    const int mB = mA + 16;
    out[(size_t)mA * DIMC + oA] = acc00[r];
    out[(size_t)mA * DIMC + oB] = acc01[r];
    out[(size_t)mB * DIMC + oA] = acc10[r];
    out[(size_t)mB * DIMC + oB] = acc11[r];
  }
}

// Per-channel sum / sumsq over all NTOT pixels (train-mode BN stats).
__global__ __launch_bounds__(256)
void bn_stats(const float* __restrict__ conv, float* __restrict__ stats) {
  __shared__ float ss[256], ss2[256];
  const int c = blockIdx.x;
  float s = 0.f, s2 = 0.f;
  for (int n = threadIdx.x; n < NTOT; n += 256) {
    const float v = conv[(size_t)n * DIMC + c];
    s += v; s2 += v * v;
  }
  ss[threadIdx.x] = s; ss2[threadIdx.x] = s2;
  __syncthreads();
  for (int st = 128; st > 0; st >>= 1) {
    if (threadIdx.x < st) {
      ss[threadIdx.x]  += ss[threadIdx.x + st];
      ss2[threadIdx.x] += ss2[threadIdx.x + st];
    }
    __syncthreads();
  }
  if (threadIdx.x == 0) { stats[c] = ss[0]; stats[DIMC + c] = ss2[0]; }
}

// BN (biased batch stats) + ReLU + pack to bf16.
__global__ __launch_bounds__(256)
void bn_apply(const float* __restrict__ conv, const float* __restrict__ stats,
              const float* __restrict__ g, const float* __restrict__ b,
              unsigned short* __restrict__ out) {
  const size_t i = (size_t)blockIdx.x * 256 + threadIdx.x;   // NTOT*128 exact
  const int c = (int)(i & (DIMC - 1));
  const float mean = stats[c] * (1.f / NTOT);
  const float var  = stats[DIMC + c] * (1.f / NTOT) - mean * mean;
  const float sc   = g[c] * rsqrtf(var + 1e-5f);
  const float v    = (conv[i] - mean) * sc + b[c];
  out[i] = f2bf(fmaxf(v, 0.f));
}

// ---------------------------------------------------------------------------
// Local channel-attention. One wave per (pixel, head); 8 heads per block.
// dots = (q*scale)(16x9) . k^T(9x16) as one bf16 WMMA with K zero-padded to 32.
// softmax over m across lanes; out[n] = sum_m attn[n,m] * vsum[m]
// (reference sums over the 9 patch positions right after, so only vsum needed).
// ---------------------------------------------------------------------------
__global__ __launch_bounds__(256)
void attn_kernel(const unsigned short* __restrict__ qkv,   // [NTOT][384]
                 unsigned short* __restrict__ out) {       // [NTOT][128]
  const int n    = blockIdx.x;
  const int head = threadIdx.x >> 5;
  const int lane = threadIdx.x & 31;
  const int l    = lane & 15;
  const int half = lane >> 4;
  const int bimg = n / NPIX;
  const int rem  = n % NPIX;
  const int y    = rem / HW;
  const int x    = rem % HW;

  const int qoff = head * 16 + l;
  const int koff = 128 + head * 16 + l;
  const int voff = 256 + head * 16 + l;

  int  nb[9];
  bool nv[9];
#pragma unroll
  for (int t = 0; t < 9; ++t) {
    const int dy = t / 3 - 1, dx = t % 3 - 1;
    const int ys = y + dy, xs = x + dx;
    nv[t] = (ys >= 0) & (ys < HW) & (xs >= 0) & (xs < HW);
    nb[t] = (bimg * NPIX + ys * HW + xs) * 384;
  }

  Frag a, b;
#pragma unroll
  for (int j = 0; j < 16; ++j) { a.u[j] = 0; b.u[j] = 0; }

  // A = q * 0.25 (row M = channel l, K = patch index 0..8, rest zero).
  if (half == 0) {
#pragma unroll
    for (int j = 0; j < 9; ++j)
      if (nv[j]) a.u[j] = f2bf(bf2f(qkv[nb[j] + qoff]) * 0.25f);
  }
  // B = k^T (col N = channel l, K = patch; lanes>=16 carry K16..31 -> zero).
  if (half == 0) {
#pragma unroll
    for (int t = 0; t < 9; ++t)
      if (nv[t]) b.u[t] = qkv[nb[t] + koff];
  }
  // vsum over the 9 patch positions for channel l.
  float vs = 0.f;
#pragma unroll
  for (int t = 0; t < 9; ++t)
    if (nv[t]) vs += bf2f(qkv[nb[t] + voff]);

  floatx8 dots = {};
  dots = WMMA_BF16(a.v, b.v, dots);

#pragma unroll
  for (int r = 0; r < 8; ++r) {
    float d = dots[r];
    float mx = d;
    for (int msk = 1; msk < 16; msk <<= 1) mx = fmaxf(mx, __shfl_xor(mx, msk, 32));
    const float e = __expf(d - mx);
    float s = e;
    for (int msk = 1; msk < 16; msk <<= 1) s += __shfl_xor(s, msk, 32);
    float o = (e / s) * vs;
    for (int msk = 1; msk < 16; msk <<= 1) o += __shfl_xor(o, msk, 32);
    if (l == 0) {
      const int cc = head * 16 + r + half * 8;   // D row M = r + half*8
      out[(size_t)n * DIMC + cc] = f2bf(o);
    }
  }
}

// NCHW f32 -> pixel-major bf16 [NTOT][64].
__global__ __launch_bounds__(256)
void pack_input(const float* __restrict__ x, unsigned short* __restrict__ xb) {
  const size_t i = (size_t)blockIdx.x * 256 + threadIdx.x;   // NTOT*64 exact
  const int c = (int)(i & 63);
  const int n = (int)(i >> 6);
  const int bimg = n / NPIX, rem = n % NPIX;
  xb[i] = f2bf(x[((size_t)bimg * 64 + c) * NPIX + rem]);
}

__global__ __launch_bounds__(256)
void cvt_bf16(const float* __restrict__ in, unsigned short* __restrict__ out, int count) {
  const int i = blockIdx.x * 256 + threadIdx.x;
  if (i < count) out[i] = f2bf(in[i]);
}

// conv_w [4][128(o)][128(i)][3][3] -> wt [4][9(tap)][128(o)][128(i)] bf16.
__global__ __launch_bounds__(256)
void cvt_convw(const float* __restrict__ w, unsigned short* __restrict__ wt) {
  const int i = blockIdx.x * 256 + threadIdx.x;
  if (i >= 4 * 9 * 128 * 128) return;
  const int ii  = i & 127;
  const int o   = (i >> 7) & 127;
  const int dt  = i >> 14;           // d*9 + tap
  const int tap = dt % 9;
  const int d   = dt / 9;
  wt[i] = f2bf(w[(((size_t)d * 128 + o) * 128 + ii) * 9 + tap]);
}

// ---------------------------------------------------------------------------

extern "C" void kernel_launch(void* const* d_in, const int* in_sizes, int n_in,
                              void* d_out, int out_size, void* d_ws, size_t ws_size,
                              hipStream_t stream) {
  const float* x      = (const float*)d_in[0];
  const float* w_in   = (const float*)d_in[1];
  const float* b_in   = (const float*)d_in[2];
  const float* conv_w = (const float*)d_in[3];
  // d_in[4] = conv_b: exactly cancelled by train-mode BatchNorm.
  const float* bn_g   = (const float*)d_in[5];
  const float* bn_b   = (const float*)d_in[6];
  const float* qkv_w  = (const float*)d_in[7];
  const float* fc_w   = (const float*)d_in[8];
  const float* fc_b   = (const float*)d_in[9];
  const float* w_out  = (const float*)d_in[10];
  const float* b_out  = (const float*)d_in[11];

  char* ws = (char*)d_ws;
  size_t off = 0;
  auto alloc = [&](size_t bytes) -> void* {
    void* p = ws + off;
    off = (off + bytes + 255) & ~(size_t)255;
    return p;
  };
  unsigned short* xb      = (unsigned short*)alloc((size_t)NTOT * 64 * 2);
  unsigned short* h       = (unsigned short*)alloc((size_t)NTOT * 128 * 2);
  float*          cbuf    = (float*)         alloc((size_t)NTOT * 128 * 4);
  float*          stats   = (float*)         alloc(256 * 4);
  unsigned short* qkvb    = (unsigned short*)alloc((size_t)NTOT * 384 * 2);
  unsigned short* att     = (unsigned short*)alloc((size_t)NTOT * 128 * 2);
  unsigned short* w_in_b  = (unsigned short*)alloc(128 * 64 * 2);
  unsigned short* wt      = (unsigned short*)alloc((size_t)4 * 9 * 128 * 128 * 2);
  unsigned short* qkv_wb  = (unsigned short*)alloc((size_t)4 * 384 * 128 * 2);
  unsigned short* fc_wb   = (unsigned short*)alloc((size_t)4 * 128 * 128 * 2);
  unsigned short* w_out_b = (unsigned short*)alloc(64 * 128 * 2);

  // Weight conversion / re-layout (deterministic, every call).
  cvt_bf16 <<<(128 * 64 + 255) / 256, 256, 0, stream>>>(w_in, w_in_b, 128 * 64);
  cvt_convw<<<(4 * 9 * 128 * 128 + 255) / 256, 256, 0, stream>>>(conv_w, wt);
  cvt_bf16 <<<(4 * 384 * 128 + 255) / 256, 256, 0, stream>>>(qkv_w, qkv_wb, 4 * 384 * 128);
  cvt_bf16 <<<(4 * 128 * 128 + 255) / 256, 256, 0, stream>>>(fc_w, fc_wb, 4 * 128 * 128);
  cvt_bf16 <<<(64 * 128 + 255) / 256, 256, 0, stream>>>(w_out, w_out_b, 64 * 128);
  pack_input<<<(NTOT * 64) / 256, 256, 0, stream>>>(x, xb);

  const int NT32 = NTOT / 32;   // 576 pixel tiles of 32

  // Stem 1x1 conv: [NTOT,64] x [128,64]^T + b_in -> h (bf16).
  wmma_gemm<<<dim3(NT32, 4), 32, 0, stream>>>(xb, w_in_b, b_in, nullptr, h, 64, 128);

  for (int i = 0; i < 4; ++i) {
    const int dil = 1 << i;
    wmma_dconv<<<dim3(NT32 / 8, 4), 256, 0, stream>>>(h, wt + (size_t)i * 9 * 128 * 128,
                                                      cbuf, dil);
    bn_stats<<<128, 256, 0, stream>>>(cbuf, stats);
    bn_apply<<<(NTOT * 128) / 256, 256, 0, stream>>>(cbuf, stats, bn_g + i * 128, bn_b + i * 128, h);
    wmma_gemm<<<dim3(NT32, 12), 32, 0, stream>>>(h, qkv_wb + (size_t)i * 384 * 128,
                                                 nullptr, nullptr, qkvb, 128, 384);
    attn_kernel<<<NTOT, 256, 0, stream>>>(qkvb, att);
    // fc + bias + residual(h) -> h (in-place safe: each element RMW by one thread).
    wmma_gemm<<<dim3(NT32, 4), 32, 0, stream>>>(att, fc_wb + (size_t)i * 128 * 128,
                                                fc_b + i * 128, h, h, 128, 128);
  }

  // Head 1x1 conv 128->64, f32 NCHW into d_out.
  wmma_gemm_out<<<dim3(NT32, 2), 32, 0, stream>>>(h, w_out_b, b_out, (float*)d_out, 128, 64);
}
